// LSTM_38714835206297
// MI455X (gfx1250) — matrix operand order repaired and checked by reading, not compiled, not run on hardware
//
#include <hip/hip_runtime.h>
#include <hip/hip_bf16.h>

// ---------------------------------------------------------------------------
// Types for CDNA5 WMMA (wave32): v_wmma_f32_16x16x32_bf16
// ---------------------------------------------------------------------------
typedef __bf16 bf16_t;
typedef __attribute__((ext_vector_type(16))) __bf16 v16bf;
typedef __attribute__((ext_vector_type(8)))  __bf16 v8bf;
typedef __attribute__((ext_vector_type(8)))  float  v8f;
typedef __attribute__((ext_vector_type(8)))  float  v8ff;

#define N_BATCH 64
#define SEQ_L   256
#define IN_I    256
#define HID_H   1024
#define GATES4H 4096
#define DEEP_D  3
#define NH      (N_BATCH * HID_H)
// Persistent recurrent kernel grid: (H/128) x (N/16) = 8 x 4 = 32 blocks.
#define RB_X    (HID_H / 128)
#define RB_Y    (N_BATCH / 16)
#define NBLOCKS (RB_X * RB_Y)

static __device__ __forceinline__ v8f wmma_bf16(v16bf a, v16bf b, v8f c) {
  // 8 args: (neg_a, A, neg_b, B, c_mod, C, reuse_a, reuse_b)
  return __builtin_amdgcn_wmma_f32_16x16x32_bf16(false, a, false, b, (short)0, c,
                                                 false, false);
}

// A-matrix fragment (16x32, M x K), bf16 row-major source, leading dim `ld`.
// ISA layout: lanes 0-15 -> row M=lane, K = {k0..k0+7} in v0..3, {k0+16..23} in v4..7
//             lanes 16-31 -> row M=lane-16, K = {k0+8..15} and {k0+24..31}
static __device__ __forceinline__ v16bf load_fragA(const bf16_t* base,
                                                   size_t ld, int k0) {
  int lane = threadIdx.x & 31;
  int r = lane & 15, hh = lane >> 4;
  const bf16_t* p = base + (size_t)r * ld + (size_t)(k0 + hh * 8);
  v8bf lo = *(const v8bf*)(p);        // K k0+8h .. k0+8h+7
  v8bf hi = *(const v8bf*)(p + 16);   // K k0+16+8h .. k0+16+8h+7
  return __builtin_shufflevector(lo, hi, 0,1,2,3,4,5,6,7,8,9,10,11,12,13,14,15);
}

// Same as load_fragA but source is fp32 (for the raw input x), converted to bf16.
static __device__ __forceinline__ v16bf load_fragA_f32(const float* __restrict__ base,
                                                       size_t ld, int k0) {
  int lane = threadIdx.x & 31;
  int r = lane & 15, hh = lane >> 4;
  const float* p = base + (size_t)r * ld + (size_t)(k0 + hh * 8);
  v8ff lo = *(const v8ff*)(p);
  v8ff hi = *(const v8ff*)(p + 16);
  v16bf o;
#pragma unroll
  for (int i = 0; i < 8; ++i) { o[i] = (bf16_t)lo[i]; o[i + 8] = (bf16_t)hi[i]; }
  return o;
}

// B-matrix fragment (32x16, K x N). B = W^T, W row-major [n][k], leading dim `ld`.
// Layout (matches A): lanes 0-15 hold col N=lane, K = {k0..k0+7} and {k0+16..23};
//                     lanes 16-31 hold col N=lane-16, K = {k0+8..15} and {k0+24..31}.
static __device__ __forceinline__ v16bf load_fragB(const bf16_t* __restrict__ wbase,
                                                   size_t ld, int k0) {
  int lane = threadIdx.x & 31;
  int n = lane & 15, hh = lane >> 4;
  const bf16_t* p = wbase + (size_t)n * ld + (size_t)(k0 + hh * 8);
  v8bf lo = *(const v8bf*)(p);
  v8bf hi = *(const v8bf*)(p + 16);
  return __builtin_shufflevector(lo, hi, 0,1,2,3,4,5,6,7,8,9,10,11,12,13,14,15);
}

static __device__ __forceinline__ float sigf(float x) {
  return 1.0f / (1.0f + __expf(-x));
}

// ---------------------------------------------------------------------------
// Prep: bf16-convert W_ih0; fold (W_ih+W_hh)->bf16 and biases; zero c + barrier.
// ---------------------------------------------------------------------------
__global__ void prep_all(const float* __restrict__ W_ih0,
                         const float* __restrict__ b_ih0, const float* __restrict__ b_hh0,
                         const float* __restrict__ W_ih,  const float* __restrict__ W_hh,
                         const float* __restrict__ b_ih,  const float* __restrict__ b_hh,
                         bf16_t* __restrict__ W0b, float* __restrict__ b0s,
                         bf16_t* __restrict__ Wcb, float* __restrict__ bcs,
                         float* __restrict__ cst, unsigned* __restrict__ bar) {
  size_t tid = (size_t)blockIdx.x * blockDim.x + threadIdx.x;
  size_t stride = (size_t)gridDim.x * blockDim.x;
  if (tid == 0) *bar = 0u;
  for (size_t i = tid; i < (size_t)GATES4H * IN_I; i += stride)
    W0b[i] = (bf16_t)W_ih0[i];
  for (size_t i = tid; i < (size_t)DEEP_D * GATES4H * HID_H; i += stride)
    Wcb[i] = (bf16_t)(W_ih[i] + W_hh[i]);
  for (size_t i = tid; i < (size_t)GATES4H; i += stride)
    b0s[i] = b_ih0[i] + b_hh0[i];
  for (size_t i = tid; i < (size_t)DEEP_D * GATES4H; i += stride)
    bcs[i] = b_ih[i] + b_hh[i];
  for (size_t i = tid; i < (size_t)DEEP_D * NH; i += stride)
    cst[i] = 0.0f;
}

// ---------------------------------------------------------------------------
// Layer 0: non-recurrent (h=c=0 every step in the reference).
// gates = x_t @ W_ih0^T + (b_ih0+b_hh0);  h0[t] = sig(o)*tanh(sig(i)*tanh(g)).
// All 256 timesteps in parallel. grid = (H/128, N/16, L), block = 256 (8 waves);
// each wave owns one 16x16 hidden tile across all 4 gate blocks, so the
// activation fuses entirely in-register (no gate matrix round-trip).
// ---------------------------------------------------------------------------
__global__ __launch_bounds__(256)
void lstm_layer0(const float* __restrict__ x,     // (N, L, I) fp32
                 const bf16_t* __restrict__ W0,   // (4H, I) bf16
                 const float* __restrict__ bias0, // (4H)
                 bf16_t* __restrict__ h0all) {    // (L, N, H) bf16
  const int wave = threadIdx.x >> 5;
  const int lane = threadIdx.x & 31;
  const int j0 = (blockIdx.x * 8 + wave) * 16;  // hidden column tile
  const int m0 = blockIdx.y * 16;               // batch row tile
  const int t  = blockIdx.z;

  const float* A = x + (size_t)m0 * (SEQ_L * IN_I) + (size_t)t * IN_I;
  const bf16_t* Bi = W0 + (size_t)(0 * HID_H + j0) * IN_I;
  const bf16_t* Bf = W0 + (size_t)(1 * HID_H + j0) * IN_I;
  const bf16_t* Bg = W0 + (size_t)(2 * HID_H + j0) * IN_I;
  const bf16_t* Bo = W0 + (size_t)(3 * HID_H + j0) * IN_I;

  v8f ai = {}, af = {}, ag = {}, ao = {};
  for (int k0 = 0; k0 < IN_I; k0 += 32) {
    v16bf a = load_fragA_f32(A, (size_t)SEQ_L * IN_I, k0);
    ai = wmma_bf16(a, load_fragB(Bi, IN_I, k0), ai);
    af = wmma_bf16(a, load_fragB(Bf, IN_I, k0), af);
    ag = wmma_bf16(a, load_fragB(Bg, IN_I, k0), ag);
    ao = wmma_bf16(a, load_fragB(Bo, IN_I, k0), ao);
  }

  const int col   = j0 + (lane & 15);
  const int rbase = m0 + ((lane >> 4) << 3);
  const float bi_ = bias0[0 * HID_H + col];
  const float bg_ = bias0[2 * HID_H + col];
  const float bo_ = bias0[3 * HID_H + col];
  bf16_t* hout = h0all + (size_t)t * NH;
  (void)af;
#pragma unroll
  for (int r = 0; r < 8; ++r) {
    const int row = rbase + r;
    const float iv = sigf(ai[r] + bi_);
    const float gv = tanhf(ag[r] + bg_);
    const float ov = sigf(ao[r] + bo_);
    const float cn = iv * gv;                 // c_prev == 0: f-gate irrelevant
    hout[(size_t)row * HID_H + col] = (bf16_t)(ov * tanhf(cn));
  }
}

// ---------------------------------------------------------------------------
// Persistent recurrent kernel: all 256 timesteps x 3 layers in ONE launch.
// 32 co-resident blocks; monotonic L2-atomic grid barrier between stages
// replaces 768 kernel launches (launch latency >> per-stage work on MI455X:
// each stage is 0.54 GFLOP + 8MB of L2-resident weights).
//
// Per stage: gates = h_in @ (W_ih+W_hh)^T + b (folded: reference passes the
// same h as both x and hidden); c = sig(f)*c + sig(i)*tanh(g);
// h_out = sig(o)*tanh(c). Each block stages its 16-row A panel (32KB) in LDS
// once per stage (8 waves share it); each wave owns one 16x16 hidden tile
// across the 4 gate blocks -> LSTM cell update fused in-register.
// c tiles are block-private across all stages (no cross-block ordering needed).
// ---------------------------------------------------------------------------
__global__ __launch_bounds__(256)
void lstm_recurrent(const bf16_t* __restrict__ h0all, // (L, N, H) bf16
                    const bf16_t* __restrict__ Wcb,   // (3, 4H, H) bf16 folded
                    const float* __restrict__ bcs,    // (3, 4H) folded
                    float* __restrict__ cst,          // (3, N, H) fp32 state
                    bf16_t* __restrict__ hbuf,        // (3, N, H) bf16
                    unsigned* bar) {
  __shared__ bf16_t sA[16 * HID_H];                   // 32 KB A panel

  const int wave = threadIdx.x >> 5;
  const int lane = threadIdx.x & 31;
  const int j0 = (blockIdx.x * 8 + wave) * 16;
  const int m0 = blockIdx.y * 16;
  const int col   = j0 + (lane & 15);
  const int rbase = m0 + ((lane >> 4) << 3);

  // Time-invariant per-lane gate biases (12 scalars), hoisted out of the loop.
  float bI[DEEP_D], bF[DEEP_D], bG[DEEP_D], bO[DEEP_D];
#pragma unroll
  for (int l = 0; l < DEEP_D; ++l) {
    bI[l] = bcs[l * GATES4H + 0 * HID_H + col];
    bF[l] = bcs[l * GATES4H + 1 * HID_H + col];
    bG[l] = bcs[l * GATES4H + 2 * HID_H + col];
    bO[l] = bcs[l * GATES4H + 3 * HID_H + col];
  }

  unsigned step = 0;
  for (int t = 0; t < SEQ_L; ++t) {
    for (int l = 0; l < DEEP_D; ++l) {
      const bf16_t* hin = (l == 0) ? (h0all + (size_t)t * NH)
                                   : (hbuf + (size_t)(l - 1) * NH);
      const bf16_t* W = Wcb + (size_t)l * GATES4H * HID_H;
      float* c        = cst  + (size_t)l * NH;
      bf16_t* hout    = hbuf + (size_t)l * NH;

      // Stage the block's 16-row A panel (contiguous 32KB) into LDS.
      {
        const v8bf* src = (const v8bf*)(hin + (size_t)m0 * HID_H);
        v8bf* dst = (v8bf*)sA;
        for (int i = threadIdx.x; i < (16 * HID_H) / 8; i += 256)
          dst[i] = src[i];
      }
      __syncthreads();

      const bf16_t* Bi = W + (size_t)(0 * HID_H + j0) * HID_H;
      const bf16_t* Bf = W + (size_t)(1 * HID_H + j0) * HID_H;
      const bf16_t* Bg = W + (size_t)(2 * HID_H + j0) * HID_H;
      const bf16_t* Bo = W + (size_t)(3 * HID_H + j0) * HID_H;

      v8f ai = {}, af = {}, ag = {}, ao = {};
      for (int k0 = 0; k0 < HID_H; k0 += 32) {
        v16bf a = load_fragA(sA, HID_H, k0);      // ds_load from LDS
        ai = wmma_bf16(a, load_fragB(Bi, HID_H, k0), ai);
        af = wmma_bf16(a, load_fragB(Bf, HID_H, k0), af);
        ag = wmma_bf16(a, load_fragB(Bg, HID_H, k0), ag);
        ao = wmma_bf16(a, load_fragB(Bo, HID_H, k0), ao);
      }

#pragma unroll
      for (int r = 0; r < 8; ++r) {
        const int row = rbase + r;
        const size_t idx = (size_t)row * HID_H + col;
        const float iv = sigf(ai[r] + bI[l]);
        const float fv = sigf(af[r] + bF[l]);
        const float gv = tanhf(ag[r] + bG[l]);
        const float ov = sigf(ao[r] + bO[l]);
        const float cn = fv * c[idx] + iv * gv;
        c[idx] = cn;
        hout[idx] = (bf16_t)(ov * tanhf(cn));
      }

      // Grid-wide release/acquire barrier (monotonic counter, zeroed by prep).
      ++step;
      __syncthreads();
      __threadfence();                        // release: publish hout stores
      if (threadIdx.x == 0) {
        __hip_atomic_fetch_add(bar, 1u, __ATOMIC_RELEASE, __HIP_MEMORY_SCOPE_AGENT);
        const unsigned target = step * NBLOCKS;
        while (__hip_atomic_load(bar, __ATOMIC_ACQUIRE, __HIP_MEMORY_SCOPE_AGENT) < target)
          __builtin_amdgcn_s_sleep(2);
      }
      __syncthreads();
      __threadfence();                        // acquire: invalidate stale lines
    }
  }
}

// ---------------------------------------------------------------------------
// Output projection: out[n] = h3[n,:] . W_out + b_out   (64 x 1)
// ---------------------------------------------------------------------------
__global__ void final_out(const bf16_t* __restrict__ h3,
                          const float* __restrict__ W_out,
                          const float* __restrict__ b_out,
                          float* __restrict__ out) {
  const int n = threadIdx.x;
  if (n < N_BATCH) {
    float s = 0.0f;
    for (int j = 0; j < HID_H; ++j)
      s += (float)h3[(size_t)n * HID_H + j] * W_out[j];
    out[n] = s + b_out[0];
  }
}

// ---------------------------------------------------------------------------
// Host launcher: 4 launches total (prep -> layer0 -> persistent -> out).
// ---------------------------------------------------------------------------
extern "C" void kernel_launch(void* const* d_in, const int* in_sizes, int n_in,
                              void* d_out, int out_size, void* d_ws, size_t ws_size,
                              hipStream_t stream) {
  (void)in_sizes; (void)n_in; (void)out_size; (void)ws_size;
  const float* x     = (const float*)d_in[0];   // (64,256,256)
  const float* W_ih0 = (const float*)d_in[1];   // (4096,256)
  // d_in[2] = W_hh0 — mathematically dead (h==0 for layer 0)
  const float* b_ih0 = (const float*)d_in[3];   // (4096)
  const float* b_hh0 = (const float*)d_in[4];   // (4096)
  const float* W_ih  = (const float*)d_in[5];   // (3,4096,1024)
  const float* W_hh  = (const float*)d_in[6];   // (3,4096,1024)
  const float* b_ih  = (const float*)d_in[7];   // (3,4096)
  const float* b_hh  = (const float*)d_in[8];   // (3,4096)
  const float* W_out = (const float*)d_in[9];   // (1,1024)
  const float* b_out = (const float*)d_in[10];  // (1)
  float* out = (float*)d_out;                   // (64)

  // Workspace layout (all offsets 256B-aligned). Total ~59 MB.
  char* w = (char*)d_ws;
  bf16_t* W0b   = (bf16_t*)w;  w += (size_t)GATES4H * IN_I * 2;             // 2 MB
  bf16_t* Wcb   = (bf16_t*)w;  w += (size_t)DEEP_D * GATES4H * HID_H * 2;   // 24 MB
  float*  b0s   = (float*)w;   w += (size_t)GATES4H * 4;                    // 16 KB
  float*  bcs   = (float*)w;   w += (size_t)DEEP_D * GATES4H * 4;           // 48 KB
  float*  cst   = (float*)w;   w += (size_t)DEEP_D * NH * 4;                // 768 KB
  bf16_t* h0all = (bf16_t*)w;  w += (size_t)SEQ_L * NH * 2;                 // 32 MB
  bf16_t* hbuf  = (bf16_t*)w;  w += (size_t)DEEP_D * NH * 2;                // 384 KB
  unsigned* bar = (unsigned*)w; w += 256;

  prep_all<<<4096, 256, 0, stream>>>(W_ih0, b_ih0, b_hh0, W_ih, W_hh, b_ih, b_hh,
                                     W0b, b0s, Wcb, bcs, cst, bar);

  // Layer 0 for all timesteps in parallel (non-recurrent).
  lstm_layer0<<<dim3(HID_H / 128, N_BATCH / 16, SEQ_L), 256, 0, stream>>>(
      x, W0b, b0s, h0all);

  // Entire 256-step x 3-layer recurrence in one persistent launch.
  lstm_recurrent<<<dim3(RB_X, RB_Y), 256, 0, stream>>>(
      h0all, Wcb, bcs, cst, hbuf, bar);

  // Last layer's hidden at final timestep -> (64,1) projection.
  final_out<<<1, 64, 0, stream>>>(hbuf + (size_t)(DEEP_D - 1) * NH, W_out, b_out, out);
}
